// PartBlock_OneFC_82995948028522
// MI455X (gfx1250) — matrix operand order repaired and checked by reading, not compiled, not run on hardware
//
#include <hip/hip_runtime.h>
#include <hip/hip_bf16.h>

// ---- types for WMMA fragments ------------------------------------------------
typedef __attribute__((ext_vector_type(16))) __bf16 v16bf;
typedef __attribute__((ext_vector_type(8)))  __bf16 v8bf;
typedef __attribute__((ext_vector_type(8)))  float  v8f;
typedef __attribute__((ext_vector_type(4)))  float  v4f;

#define P_PARTS 62
#define NROWS   8192
#define CIN     128
#define COUT    256
#define LDST    136          // LDS row stride in bf16 elems (128 + 8 pad -> no bank conflicts)
#define ROWS_PER_BLOCK 128   // 8 waves x 16 rows
#define LN_EPS  1e-5f

// bf16 by truncation (cheap, exact hi-part for split-GEMM)
__device__ __forceinline__ __bf16 bf_trunc(float f) {
    unsigned short s = (unsigned short)(__float_as_uint(f) >> 16);
    return __builtin_bit_cast(__bf16, s);
}
__device__ __forceinline__ float bf_to_f(__bf16 h) {
    unsigned u = ((unsigned)__builtin_bit_cast(unsigned short, h)) << 16;
    return __uint_as_float(u);
}

extern "C" __global__ __launch_bounds__(256)
void partblock_fc_ln_kernel(const float* __restrict__ x,
                            const float* __restrict__ W,
                            const float* __restrict__ gamma,
                            const float* __restrict__ beta,
                            float* __restrict__ out)
{
    extern __shared__ __bf16 smem[];
    __bf16* sWhi = smem;                       // [COUT][LDST] bf16 hi part of W[p]
    __bf16* sWlo = smem + COUT * LDST;         // [COUT][LDST] bf16 lo part of W[p]
    float*  sG   = (float*)(smem + 2 * COUT * LDST);
    float*  sB   = sG + COUT;

    const int p   = blockIdx.y;
    const int tid = threadIdx.x;

    // ---- stage W[p] (256x128 fp32, L2-resident) into LDS as hi/lo bf16 ------
    {
        const float* wrow = W + ((size_t)p * COUT + tid) * CIN;  // tid == output channel d
        __bf16* dhi = sWhi + tid * LDST;
        __bf16* dlo = sWlo + tid * LDST;
        #pragma unroll
        for (int c = 0; c < CIN; c += 4) {
            v4f wv = *(const v4f*)(wrow + c);
            #pragma unroll
            for (int j = 0; j < 4; ++j) {
                __bf16 hi = bf_trunc(wv[j]);
                dhi[c + j] = hi;
                dlo[c + j] = bf_trunc(wv[j] - bf_to_f(hi));
            }
        }
        sG[tid] = gamma[tid];
        sB[tid] = beta[tid];
    }
    __syncthreads();

    // ---- per-wave 16-row strip, all 256 output cols --------------------------
    const int  wave    = tid >> 5;
    const int  lane    = tid & 31;
    const int  lc      = lane & 15;
    const int  half    = lane >> 4;            // 0: lanes 0-15, 1: lanes 16-31
    const int  nbase   = blockIdx.x * ROWS_PER_BLOCK + wave * 16;
    const int  arow    = nbase + lc;           // A row supplied by this lane
    const int  aklo    = half ? 8  : 0;        // A layout: K sub-offset per lane half
    const int  bkoff   = half ? 8  : 0;        // B layout: K block per lane half

    v8f acc[16];
    #pragma unroll
    for (int t = 0; t < 16; ++t) {
        v8f z = {};
        acc[t] = z;
    }

    const float* xrow = x + ((size_t)p * NROWS + arow) * CIN;

    #pragma unroll
    for (int ks = 0; ks < 4; ++ks) {           // K = 128 in 4 steps of 32
        // A fragment 16x32 (ISA 16-bit A layout): elems 0-7 -> K=k0+aklo+i,
        // elems 8-15 -> K=k0+16+aklo+(i-8)
        const float* ap = xrow + ks * 32 + aklo;
        v4f f0 = *(const v4f*)(ap +  0);
        v4f f1 = *(const v4f*)(ap +  4);
        v4f f2 = *(const v4f*)(ap + 16);
        v4f f3 = *(const v4f*)(ap + 20);
        float fa[16] = { f0[0],f0[1],f0[2],f0[3], f1[0],f1[1],f1[2],f1[3],
                         f2[0],f2[1],f2[2],f2[3], f3[0],f3[1],f3[2],f3[3] };
        v16bf ahi, alo;
        #pragma unroll
        for (int i = 0; i < 16; ++i) {
            __bf16 h = bf_trunc(fa[i]);
            ahi[i] = h;
            alo[i] = bf_trunc(fa[i] - bf_to_f(h));
        }

        #pragma unroll
        for (int t = 0; t < 16; ++t) {         // 16 column tiles of 16 channels
            const int col  = t * 16 + lc;
            const int boff = col * LDST + ks * 32 + bkoff;
            v8bf h0 = *(const v8bf*)(sWhi + boff);
            v8bf h1 = *(const v8bf*)(sWhi + boff + 16);
            v8bf l0 = *(const v8bf*)(sWlo + boff);
            v8bf l1 = *(const v8bf*)(sWlo + boff + 16);
            v16bf bh = __builtin_shufflevector(h0, h1, 0,1,2,3,4,5,6,7,8,9,10,11,12,13,14,15);
            v16bf bl = __builtin_shufflevector(l0, l1, 0,1,2,3,4,5,6,7,8,9,10,11,12,13,14,15);

            // split-GEMM: hi*hi + hi*lo + lo*hi  (f32 accumulate)
            acc[t] = __builtin_amdgcn_wmma_f32_16x16x32_bf16(
                         false, ahi, false, bh, (short)0, acc[t], false, false);
            acc[t] = __builtin_amdgcn_wmma_f32_16x16x32_bf16(
                         false, ahi, false, bl, (short)0, acc[t], false, false);
            acc[t] = __builtin_amdgcn_wmma_f32_16x16x32_bf16(
                         false, alo, false, bh, (short)0, acc[t], false, false);
        }
    }

    // ---- fused LayerNorm over the 256 channels + store ----------------------
    // C/D layout: VGPR r, lanes 0-15 -> row M=r, N=lane; lanes 16-31 -> M=8+r.
    #pragma unroll
    for (int r = 0; r < 8; ++r) {
        float s = 0.f, q = 0.f;
        #pragma unroll
        for (int t = 0; t < 16; ++t) { float v = acc[t][r]; s += v; q += v * v; }
        // reduce across the 16 lanes of each half-wave (xor masks < 16 stay in-half)
        #pragma unroll
        for (int m = 1; m < 16; m <<= 1) {
            s += __shfl_xor(s, m, 32);
            q += __shfl_xor(q, m, 32);
        }
        const float mean = s * (1.0f / COUT);
        const float var  = q * (1.0f / COUT) - mean * mean;
        const float rstd = rsqrtf(var + LN_EPS);

        const int n = nbase + half * 8 + r;
        float* orow = out + ((size_t)p * NROWS + n) * COUT;
        #pragma unroll
        for (int t = 0; t < 16; ++t) {
            const int col = t * 16 + lc;
            orow[col] = (acc[t][r] - mean) * rstd * sG[col] + sB[col];
        }
    }
}

extern "C" void kernel_launch(void* const* d_in, const int* in_sizes, int n_in,
                              void* d_out, int out_size, void* d_ws, size_t ws_size,
                              hipStream_t stream) {
    (void)in_sizes; (void)n_in; (void)out_size; (void)d_ws; (void)ws_size;
    const float* x  = (const float*)d_in[0];
    const float* W  = (const float*)d_in[1];
    const float* g  = (const float*)d_in[2];
    const float* b  = (const float*)d_in[3];
    float* out = (float*)d_out;

    dim3 grid(NROWS / ROWS_PER_BLOCK, P_PARTS);   // 64 x 62 workgroups
    dim3 block(256);                              // 8 wave32
    size_t lds = (size_t)2 * COUT * LDST * sizeof(__bf16) + 2 * COUT * sizeof(float);
    partblock_fc_ln_kernel<<<grid, block, lds, stream>>>(x, W, g, b, out);
}
